// AttentionBlock_30502857736839
// MI455X (gfx1250) — matrix operand order repaired and checked by reading, not compiled, not run on hardware
//
#include <hip/hip_runtime.h>

typedef __attribute__((ext_vector_type(16))) _Float16 v16h;
typedef __attribute__((ext_vector_type(8)))  _Float16 v8h;
typedef __attribute__((ext_vector_type(8)))  float    v8f;

// ---------------------------------------------------------------------------
// WMMA helpers (CDNA5 wave32, 16x16x32 f16 -> f32)
// ---------------------------------------------------------------------------
__device__ __forceinline__ v8f vzero8() {
  v8f c;
#pragma unroll
  for (int t = 0; t < 8; ++t) c[t] = 0.0f;
  return c;
}

__device__ __forceinline__ v8f wmma16(v16h a, v16h b, v8f c) {
  // (neg_a, A, neg_b, B, c_mod, C, reuse_a, reuse_b)
  return __builtin_amdgcn_wmma_f32_16x16x32_f16(false, a, false, b, (short)0, c,
                                                false, false);
}

// CDNA5 async global->LDS copy, 16B per lane. ASYNCcnt-tracked.
__device__ __forceinline__ void async_ld_b128(unsigned lds_off, const void* gptr) {
  asm volatile("global_load_async_to_lds_b128 %0, %1, off"
               :: "v"(lds_off), "v"((unsigned long long)(uintptr_t)gptr)
               : "memory");
}
__device__ __forceinline__ void wait_async0() {
  asm volatile("s_wait_asynccnt 0x0" ::: "memory");
}
__device__ __forceinline__ unsigned lds_addr(const void* sptr) {
  return (unsigned)(uintptr_t)sptr;  // low 32 bits = LDS byte offset
}

// A fragment: 16(M) x 32(K) f16 from row-major source, row stride ld elements.
// Layout: lane m (0..15) / half g: halves 0..15 -> K=16g..16g+15
// (contiguous, matches B fragments).
__device__ __forceinline__ v16h ldfragA(const _Float16* base, int ld) {
  const int lane = threadIdx.x & 31;
  const int m = lane & 15, g = lane >> 4;
  const _Float16* p = base + (size_t)m * ld + 16 * g;
  v16h r;
  reinterpret_cast<uint4*>(&r)[0] = *reinterpret_cast<const uint4*>(p);
  reinterpret_cast<uint4*>(&r)[1] = *reinterpret_cast<const uint4*>(p + 8);
  return r;
}

// B fragment: 32(K) x 16(N). Source stores, for each column n, the 32 K values
// contiguously at base + n*ld + k (contiguous order used here: lane-half g holds K=16g..16g+15).
__device__ __forceinline__ v16h ldfragB(const _Float16* base, int ld) {
  const int lane = threadIdx.x & 31;
  const int n = lane & 15, g = lane >> 4;
  const _Float16* p = base + (size_t)n * ld + 16 * g;
  v16h r;
  reinterpret_cast<uint4*>(&r)[0] = *reinterpret_cast<const uint4*>(p);
  reinterpret_cast<uint4*>(&r)[1] = *reinterpret_cast<const uint4*>(p + 8);
  return r;
}

// wave32: reduce across the 16 lanes sharing bit4 (rows live per lane-half)
__device__ __forceinline__ float rmax16(float x) {
  x = fmaxf(x, __shfl_xor(x, 1, 32));
  x = fmaxf(x, __shfl_xor(x, 2, 32));
  x = fmaxf(x, __shfl_xor(x, 4, 32));
  x = fmaxf(x, __shfl_xor(x, 8, 32));
  return x;
}
__device__ __forceinline__ float radd16(float x) {
  x += __shfl_xor(x, 1, 32);
  x += __shfl_xor(x, 2, 32);
  x += __shfl_xor(x, 4, 32);
  x += __shfl_xor(x, 8, 32);
  return x;
}

// ---------------------------------------------------------------------------
// Pack a 512x512 f32 weight (row-major, [k][n]) into per-tile B-fragment
// layout: tile (kt, nt) -> 512 contiguous f16 = 32 lanes x 16 halves.
// ---------------------------------------------------------------------------
__global__ void pack_kernel(const float* __restrict__ Wq, const float* __restrict__ Wk,
                            const float* __restrict__ Wv, const float* __restrict__ Wo,
                            const float* __restrict__ W1, const float* __restrict__ W2,
                            _Float16* __restrict__ Wp) {
  const int lane = threadIdx.x & 31;
  const int tile = blockIdx.x;  // 0..511 = kt*32 + ntile
  const int mat  = blockIdx.y;  // 0..5
  const float* W = (mat == 0) ? Wq : (mat == 1) ? Wk : (mat == 2) ? Wv
                 : (mat == 3) ? Wo : (mat == 4) ? W1 : W2;
  const int kt = tile >> 5, ntile = tile & 31;
  const int g = lane >> 4;
  const int n = ntile * 16 + (lane & 15);
  _Float16* dst = Wp + ((size_t)mat * 512 + tile) * 512 + lane * 16;
#pragma unroll
  for (int j = 0; j < 16; ++j) {
    const int K = kt * 32 + 16 * g + j;
    dst[j] = (_Float16)W[(size_t)K * 512 + n];
  }
}

// ---------------------------------------------------------------------------
// LayerNorm over 512 channels, one wave per row, output f16 (ld 512).
// Either f32 input (ldIn/colOff) or f16 input (ld 512).
// ---------------------------------------------------------------------------
__global__ void ln_kernel(const float* __restrict__ inF, const _Float16* __restrict__ inH,
                          int ldIn, int colOff,
                          const float* __restrict__ gamma, const float* __restrict__ beta,
                          _Float16* __restrict__ out) {
  const int lane = threadIdx.x & 31;
  const int w = threadIdx.x >> 5;
  const int row = blockIdx.x * 8 + w;
  const int c0 = lane * 16;
  float x[16];
  if (inF) {
    const float* p = inF + (size_t)row * ldIn + colOff + c0;
#pragma unroll
    for (int j = 0; j < 16; ++j) x[j] = p[j];
  } else {
    const _Float16* p = inH + (size_t)row * 512 + c0;
#pragma unroll
    for (int j = 0; j < 16; ++j) x[j] = (float)p[j];
  }
  float s = 0.0f, sq = 0.0f;
#pragma unroll
  for (int j = 0; j < 16; ++j) { s += x[j]; sq += x[j] * x[j]; }
#pragma unroll
  for (int mask = 1; mask <= 16; mask <<= 1) {
    s  += __shfl_xor(s, mask, 32);
    sq += __shfl_xor(sq, mask, 32);
  }
  const float mean = s * (1.0f / 512.0f);
  const float var  = sq * (1.0f / 512.0f) - mean * mean;
  const float rstd = rsqrtf(var + 1e-5f);
  _Float16* op = out + (size_t)row * 512 + c0;
  const float* gp = gamma + c0;
  const float* bp = beta + c0;
#pragma unroll
  for (int j = 0; j < 16; ++j)
    op[j] = (_Float16)((x[j] - mean) * rstd * gp[j] + bp[j]);
}

__global__ void zero_kernel(uint4* __restrict__ p, int n) {
  const int idx = blockIdx.x * blockDim.x + threadIdx.x;
  if (idx < n) { uint4 z; z.x = z.y = z.z = z.w = 0u; p[idx] = z; }
}

// ---------------------------------------------------------------------------
// GEMM: C(16384x512) = epilogue( A_f16(16384x512) @ Wpacked(512x512) )
// block = 256 threads = 8 waves; wave computes one 16x64 tile; all 8 waves
// share one 64-column weight strip, staged ONCE into LDS with CDNA5 async
// global->LDS loads (ASYNCcnt), then the K-loop reads B-frags from LDS.
// grid = (128, 8). Epilogue: +bias, *scaleOut, optional relu, optional f32
// residual (ld 1024), stores f16 (ld 512) and/or f32 (ld 1024).
// ---------------------------------------------------------------------------
__global__ void gemm512_kernel(const _Float16* __restrict__ A, const _Float16* __restrict__ Wp,
                               const float* __restrict__ bias, const float* __restrict__ resid,
                               float scaleOut, int doRelu,
                               _Float16* __restrict__ outH, float* __restrict__ outF) {
  // 64 tiles (16 kt x 4 nt), 528-half padded stride to spread LDS banks.
  __shared__ __align__(64) _Float16 sB[64 * 528];

  const int tid = threadIdx.x;
  const int lane = tid & 31;
  const int w = tid >> 5;
  const int mt = blockIdx.x * 8 + w;
  const int ntb = blockIdx.y * 4;

  // ---- async-stage the 64KB weight strip: 256 threads x 16 x 16B ----
  {
    const int tileLocal = tid >> 6;     // 0..3 within a kt-group of 4 tiles
    const int piece = tid & 63;         // 16B piece within tile
#pragma unroll
    for (int kt = 0; kt < 16; ++kt) {
      const _Float16* src = Wp + ((size_t)(kt * 32 + ntb)) * 512 + tid * 8;
      const unsigned dst =
          lds_addr(sB) + (unsigned)(((kt * 4 + tileLocal) * 528 + piece * 8) * 2);
      async_ld_b128(dst, src);
    }
    wait_async0();
    __syncthreads();
  }

  const _Float16* Abase = A + (size_t)mt * 16 * 512;
  const _Float16* sBl = sB + lane * 16;
  v8f acc[4];
#pragma unroll
  for (int nt = 0; nt < 4; ++nt) acc[nt] = vzero8();
#pragma unroll 4
  for (int kt = 0; kt < 16; ++kt) {
    const v16h a = ldfragA(Abase + kt * 32, 512);
    __builtin_prefetch(Abase + (kt + 1) * 32, 0, 3);
#pragma unroll
    for (int nt = 0; nt < 4; ++nt) {
      const v16h bf = *reinterpret_cast<const v16h*>(sBl + (size_t)(kt * 4 + nt) * 528);
      acc[nt] = wmma16(a, bf, acc[nt]);
    }
  }
  const int g = lane >> 4, n = lane & 15;
#pragma unroll
  for (int nt = 0; nt < 4; ++nt) {
    const int col = (ntb + nt) * 16 + n;
    const float bn = bias[col];
#pragma unroll
    for (int vv = 0; vv < 8; ++vv) {
      const int m = mt * 16 + 8 * g + vv;  // C/D layout: row = 8g+v
      float t = (acc[nt][vv] + bn) * scaleOut;
      if (doRelu) t = fmaxf(t, 0.0f);
      if (resid) t += resid[(size_t)m * 1024 + col];
      if (outF) outF[(size_t)m * 1024 + col] = t;
      if (outH) outH[(size_t)m * 512 + col] = (_Float16)t;
    }
  }
}

// ---------------------------------------------------------------------------
// Dilated attention, flash-style. Every group attends over exactly 1024
// gathered tokens (stride r, offset i). Unit = (batch, group, segment, head):
// 60 units, 64 query-tiles each -> grid 480 blocks x 8 waves, one 16-query
// tile per wave. KV chunk of 32 staged in LDS per block: K via async
// global->LDS b128, V transposed manually so AV B-fragments are contiguous
// b128 LDS reads. q is pre-scaled by 1/sqrt(64) in the projection GEMM.
// ---------------------------------------------------------------------------
__global__ void attn_kernel(const _Float16* __restrict__ q, const _Float16* __restrict__ k,
                            const _Float16* __restrict__ v, _Float16* __restrict__ o) {
  __shared__ __align__(64) _Float16 sK[32 * 72];      // K chunk, key-major
  __shared__ __align__(64) _Float16 sVt[64 * 40];     // V chunk, d-major (transposed)
  __shared__ __align__(64) _Float16 sP[8 * 16 * 40];  // per-wave P tile (16x32)

  const int lane = threadIdx.x & 31;
  const int w = threadIdx.x >> 5;

  const int qb = blockIdx.x & 7;
  int u = blockIdx.x >> 3;
  const int hg = u & 1;
  u >>= 1;                    // 0..29
  const int bat = u / 15;
  const int si  = u % 15;
  int grp, sg;
  if (si < 8)       { grp = 0; sg = si; }
  else if (si < 12) { grp = 1; sg = si - 8; }
  else if (si < 14) { grp = 2; sg = si - 12; }
  else              { grp = 3; sg = 0; }
  const int slen = 1024 << grp;
  const int r    = 1 << grp;
  const int head = 2 * grp + hg;
  const int tok0 = sg * slen + (grp % r);  // dilation offset = i % r
  const int rs   = r * 512;                // gathered-row stride in elements

  const size_t bbase = (size_t)bat * 8192 * 512;
  const _Float16* Qb = q + bbase + (size_t)tok0 * 512 + head * 64;
  const _Float16* Kb = k + bbase + (size_t)tok0 * 512 + head * 64;
  const _Float16* Vb = v + bbase + (size_t)tok0 * 512 + head * 64;

  const int j0 = (qb * 8 + w) * 16;
  const v16h aq0 = ldfragA(Qb + (size_t)j0 * rs, rs);       // d 0..31
  const v16h aq1 = ldfragA(Qb + (size_t)j0 * rs + 32, rs);  // d 32..63

  v8f acc[4];
#pragma unroll
  for (int nt = 0; nt < 4; ++nt) acc[nt] = vzero8();
  float mrow[8], lrow[8];
#pragma unroll
  for (int vv = 0; vv < 8; ++vv) { mrow[vv] = -1e30f; lrow[vv] = 0.0f; }

  const int srow = threadIdx.x >> 3;  // 0..31: kv row within chunk
  const int part = threadIdx.x & 7;   // 8 halves per thread
  const unsigned sKdst = lds_addr(sK) + (unsigned)((srow * 72 + part * 8) * 2);

  for (int c = 0; c < 32; ++c) {      // 32 chunks x 32 keys = 1024
    // ---- cooperative staging ----
    const size_t gRow = (size_t)(c * 32 + srow) * rs;
    async_ld_b128(sKdst, Kb + gRow + part * 8);  // K: direct global->LDS
    const v8h vv8 = *reinterpret_cast<const v8h*>(Vb + gRow + part * 8);
#pragma unroll
    for (int jj = 0; jj < 8; ++jj)
      sVt[(part * 8 + jj) * 40 + srow] = vv8[jj];
    wait_async0();
    __syncthreads();

    // ---- S = Q K^T (two 16-key subtiles) ----
    v8f s0 = vzero8(), s1 = vzero8();
    v16h kb0 = ldfragB(sK, 72);
    v16h kb1 = ldfragB(sK + 32, 72);
    s0 = wmma16(aq0, kb0, s0);
    s0 = wmma16(aq1, kb1, s0);
    v16h kb2 = ldfragB(sK + 16 * 72, 72);
    v16h kb3 = ldfragB(sK + 16 * 72 + 32, 72);
    s1 = wmma16(aq0, kb2, s1);
    s1 = wmma16(aq1, kb3, s1);

    // ---- online softmax update (rows are lane-local: m = 8g+v) ----
    const int g = lane >> 4, n = lane & 15;
    _Float16* sPw = sP + w * 16 * 40;
    float mn[8], scl[8];
#pragma unroll
    for (int vv = 0; vv < 8; ++vv) {
      float rm = rmax16(fmaxf(s0[vv], s1[vv]));
      const float mnew = fmaxf(mrow[vv], rm);
      scl[vv] = __expf(mrow[vv] - mnew);
      mrow[vv] = mnew;
      mn[vv] = mnew;
    }
#pragma unroll
    for (int vv = 0; vv < 8; ++vv) {
      const float p0 = __expf(s0[vv] - mn[vv]);
      const float p1 = __expf(s1[vv] - mn[vv]);
      sPw[(8 * g + vv) * 40 + n]      = (_Float16)p0;  // C/D -> A-frag via LDS
      sPw[(8 * g + vv) * 40 + 16 + n] = (_Float16)p1;
      const float sump = radd16(p0 + p1);
      lrow[vv] = lrow[vv] * scl[vv] + sump;
#pragma unroll
      for (int nt = 0; nt < 4; ++nt) acc[nt][vv] *= scl[vv];
    }
    asm volatile("s_wait_dscnt 0" ::: "memory");  // P stores visible to frag loads

    // ---- O += P V ----
    const v16h ap = ldfragA(sPw, 40);
#pragma unroll
    for (int nt = 0; nt < 4; ++nt) {
      const v16h bv = ldfragB(sVt + nt * 16 * 40, 40);
      acc[nt] = wmma16(ap, bv, acc[nt]);
    }
    __syncthreads();
  }

  // ---- epilogue: scatter normalized output to dilated token positions ----
  float inv[8];
#pragma unroll
  for (int vv = 0; vv < 8; ++vv) inv[vv] = 1.0f / lrow[vv];
  const int g = lane >> 4, n = lane & 15;
#pragma unroll
  for (int nt = 0; nt < 4; ++nt) {
#pragma unroll
    for (int vv = 0; vv < 8; ++vv) {
      const int j = j0 + 8 * g + vv;
      const size_t tok = (size_t)tok0 + (size_t)r * j;
      o[bbase + tok * 512 + head * 64 + nt * 16 + n] =
          (_Float16)(acc[nt][vv] * inv[vv]);
    }
  }
}

// ---------------------------------------------------------------------------
// Host orchestration
// ---------------------------------------------------------------------------
extern "C" void kernel_launch(void* const* d_in, const int* in_sizes, int n_in,
                              void* d_out, int out_size, void* d_ws, size_t ws_size,
                              hipStream_t stream) {
  const float* x    = (const float*)d_in[0];
  const float* Wq   = (const float*)d_in[1];
  const float* bq   = (const float*)d_in[2];
  const float* Wk   = (const float*)d_in[3];
  const float* bk   = (const float*)d_in[4];
  const float* Wv   = (const float*)d_in[5];
  const float* bv   = (const float*)d_in[6];
  const float* lag  = (const float*)d_in[7];
  const float* lab  = (const float*)d_in[8];
  const float* Wo   = (const float*)d_in[9];
  const float* bo   = (const float*)d_in[10];
  const float* W1   = (const float*)d_in[11];
  const float* b1   = (const float*)d_in[12];
  const float* W2   = (const float*)d_in[13];
  const float* b2   = (const float*)d_in[14];
  const float* l1g  = (const float*)d_in[15];
  const float* l1b  = (const float*)d_in[16];
  const float* l2g  = (const float*)d_in[17];
  const float* l2b  = (const float*)d_in[18];
  float* outp = (float*)d_out;

  char* ws = (char*)d_ws;
  const size_t WPE = (size_t)512 * 512;    // halfs per packed weight matrix
  _Float16* WP = (_Float16*)ws;            // 6 * WPE
  char* actBase = ws + 6 * WPE * 2;
  const size_t ACT = (size_t)16384 * 512;  // halfs per activation buffer
  _Float16* X2N = (_Float16*)(actBase + 0 * ACT * 2);
  _Float16* Qh  = (_Float16*)(actBase + 1 * ACT * 2);
  _Float16* Kh  = (_Float16*)(actBase + 2 * ACT * 2);
  _Float16* Vh  = (_Float16*)(actBase + 3 * ACT * 2);
  _Float16* Oh  = (_Float16*)(actBase + 4 * ACT * 2);
  _Float16* ONh = (_Float16*)(actBase + 5 * ACT * 2);
  _Float16* Y1N = (_Float16*)(actBase + 6 * ACT * 2);
  _Float16* Hh  = (_Float16*)(actBase + 7 * ACT * 2);

  const dim3 gGemm(128, 8);

  pack_kernel<<<dim3(512, 6), 32, 0, stream>>>(Wq, Wk, Wv, Wo, W1, W2, WP);
  // x2n = LN(x2)
  ln_kernel<<<2048, 256, 0, stream>>>(x, nullptr, 1024, 512, l1g, l1b, X2N);
  // q,k,v projections (q pre-scaled by 1/sqrt(64))
  gemm512_kernel<<<gGemm, 256, 0, stream>>>(X2N, WP + 0 * WPE, bq, nullptr, 0.125f, 0, Qh, nullptr);
  gemm512_kernel<<<gGemm, 256, 0, stream>>>(X2N, WP + 1 * WPE, bk, nullptr, 1.0f, 0, Kh, nullptr);
  gemm512_kernel<<<gGemm, 256, 0, stream>>>(X2N, WP + 2 * WPE, bv, nullptr, 1.0f, 0, Vh, nullptr);
  // dilated attention with scatter into zero-initialized o
  zero_kernel<<<4096, 256, 0, stream>>>((uint4*)Oh, 1048576);
  attn_kernel<<<480, 256, 0, stream>>>(Qh, Kh, Vh, Oh);
  // MAGNETO norm + out projection + residual x1 -> y1 (d_out cols 0..511)
  ln_kernel<<<2048, 256, 0, stream>>>(nullptr, Oh, 512, 0, lag, lab, ONh);
  gemm512_kernel<<<gGemm, 256, 0, stream>>>(ONh, WP + 3 * WPE, bo, x, 1.0f, 0, nullptr, outp);
  // FFN on LN(y1), residual x2 -> y2 (d_out cols 512..1023)
  ln_kernel<<<2048, 256, 0, stream>>>(outp, nullptr, 1024, 0, l2g, l2b, Y1N);
  gemm512_kernel<<<gGemm, 256, 0, stream>>>(Y1N, WP + 4 * WPE, b1, nullptr, 1.0f, 1, Hh, nullptr);
  gemm512_kernel<<<gGemm, 256, 0, stream>>>(Hh, WP + 5 * WPE, b2, x + 512, 1.0f, 0, nullptr, outp + 512);

  (void)in_sizes; (void)n_in; (void)out_size; (void)ws_size;
}
